// StMultiSrcPointerGen_63161789055224
// MI455X (gfx1250) — matrix operand order repaired and checked by reading, not compiled, not run on hardware
//
#include <hip/hip_runtime.h>
#include <hip/hip_bf16.h>

typedef _Float16 half_t;
typedef half_t v16h __attribute__((ext_vector_type(16)));
typedef half_t v8h  __attribute__((ext_vector_type(8)));
typedef float  v8f  __attribute__((ext_vector_type(8)));

#define D_MODELC 512
#define D_EMBC   300
#define D_EMBP   320      // padded K for vocab GEMM
#define D_FIXEDC 30000
#define D_FIXEDP 30080    // padded N (= 235 * 128)
#define D_EXTC   32000
#define BSC      8
#define LPC      256
#define JQC      48
#define NAC      32
#define TTC      48
#define ROWSC    (BSC * TTC)   // 384

// ---------------- f32 -> f16 conversion (dense) ----------------
__global__ void cvt_f32_f16(const float* __restrict__ s, half_t* __restrict__ d, int n) {
  int i = blockIdx.x * blockDim.x + threadIdx.x;
  if (i < n) d[i] = (half_t)s[i];
}

// ---------------- f32 [M,W] -> f16 [M,ldd] with zero pad of columns ----------------
__global__ void cvt_pad_rows(const float* __restrict__ s, half_t* __restrict__ d,
                             int W, int ldd, int total /* M*ldd */) {
  int i = blockIdx.x * blockDim.x + threadIdx.x;
  if (i >= total) return;
  int r = i / ldd, c = i - r * ldd;
  d[i] = (c < W) ? (half_t)s[(size_t)r * W + c] : (half_t)0.0f;
}

// ------- transpose + convert: src f32 [K,N] -> dst f16 [Npad, Kpad] (B^T, K-contig) -------
__global__ __launch_bounds__(256) void transpose_cvt(
    const float* __restrict__ src, half_t* __restrict__ dst,
    int K, int N, int Kpad) {
  __shared__ float tile[32][33];
  int n0 = blockIdx.x * 32;
  int k0 = blockIdx.y * 32;
  // coalesced read along N
  for (int i = threadIdx.y; i < 32; i += 8) {
    int k = k0 + i, n = n0 + threadIdx.x;
    tile[i][threadIdx.x] = (k < K && n < N) ? src[(size_t)k * N + n] : 0.0f;
  }
  __syncthreads();
  // coalesced write along K
  for (int i = threadIdx.y; i < 32; i += 8) {
    int n = n0 + i, k = k0 + threadIdx.x;
    dst[(size_t)n * Kpad + k] = (half_t)tile[threadIdx.x][i];
  }
}

// ---------------- WMMA f16 GEMM: C[M,ldc] = A[M,lda] @ Bt[N,ldb]^T (+bias) ----------------
// Preconditions: M multiple of 128 (grid.x = M/128), Kpad multiple of 32,
// A fully valid over [M, Kpad], Bt fully valid over [grid.y*128, Kpad].
#define GBM 128
#define GBN 128
#define GBK 32

__global__ __launch_bounds__(256) void gemm_f16_wmma(
    const half_t* __restrict__ A, const half_t* __restrict__ Bt,
    const float* __restrict__ bias, float* __restrict__ C,
    int lda, int ldb, int Kpad, int N, int ldc)
{
  __shared__ half_t As[GBM][GBK];   // [m][k]
  __shared__ half_t Bs[GBN][GBK];   // [n][k] K-contiguous

  const int tid  = threadIdx.x;
  const int lane = tid & 31;
  const int wave = tid >> 5;
  const int l16  = lane & 15;
  const int lhi  = lane >> 4;           // 0 or 1
  const int wm   = (wave & 1) * 64;     // 2 waves along M
  const int wn   = (wave >> 1) * 32;    // 4 waves along N
  const int m0   = blockIdx.x * GBM;
  const int n0   = blockIdx.y * GBN;

  const int r  = tid >> 1;              // 0..127 (row of A tile / col of B tile)
  const int kh = (tid & 1) * 16;        // 0 or 16

  v8f acc[4][2];
  for (int i = 0; i < 4; ++i)
    for (int t = 0; t < 2; ++t)
      for (int e = 0; e < 8; ++e) acc[i][t][e] = 0.0f;

  for (int k0 = 0; k0 < Kpad; k0 += GBK) {
    // prefetch tile into registers (unconditional, fully in-bounds)
    const half_t* ga = A  + (size_t)(m0 + r) * lda + k0 + kh;
    const half_t* gb = Bt + (size_t)(n0 + r) * ldb + k0 + kh;
    v8h a0 = *(const v8h*)(ga);
    v8h a1 = *(const v8h*)(ga + 8);
    v8h b0 = *(const v8h*)(gb);
    v8h b1 = *(const v8h*)(gb + 8);

    __syncthreads();                    // previous compute done
    *(v8h*)&As[r][kh]     = a0;
    *(v8h*)&As[r][kh + 8] = a1;
    *(v8h*)&Bs[r][kh]     = b0;
    *(v8h*)&Bs[r][kh + 8] = b1;
    __syncthreads();

    // B fragments: column n = wn+16t+l16, K = lhi*16 .. +15 (contiguous)
    v16h bfrag[2];
    #pragma unroll
    for (int t = 0; t < 2; ++t) {
      const half_t* bp = &Bs[wn + 16 * t + l16][lhi * 16];
      v8h lo = *(const v8h*)(bp);
      v8h hi = *(const v8h*)(bp + 8);
      #pragma unroll
      for (int j = 0; j < 8; ++j) { bfrag[t][j] = lo[j]; bfrag[t][8 + j] = hi[j]; }
    }
    // A fragments: row m = wm+16i+l16, K = lhi*16 .. +15 (contiguous)
    #pragma unroll
    for (int i = 0; i < 4; ++i) {
      const half_t* ap = &As[wm + 16 * i + l16][lhi * 16];
      v8h lo = *(const v8h*)(ap);
      v8h hi = *(const v8h*)(ap + 8);
      v16h afrag;
      #pragma unroll
      for (int j = 0; j < 8; ++j) { afrag[j] = lo[j]; afrag[8 + j] = hi[j]; }
      #pragma unroll
      for (int t = 0; t < 2; ++t) {
        acc[i][t] = __builtin_amdgcn_wmma_f32_16x16x32_f16(
            false, afrag, false, bfrag[t], (short)0, acc[i][t], false, false);
      }
    }
  }

  // store: lane<16 -> rows +0..7, lane>=16 -> rows +8..15, col = n0+wn+16t+l16
  for (int i = 0; i < 4; ++i) {
    for (int t = 0; t < 2; ++t) {
      int cn = n0 + wn + 16 * t + l16;
      if (cn >= N) continue;
      float bv = bias ? bias[cn] : 0.0f;
      int mb = m0 + wm + 16 * i + lhi * 8;
      #pragma unroll
      for (int e = 0; e < 8; ++e) {
        C[(size_t)(mb + e) * ldc + cn] = acc[i][t][e] + bv;
      }
    }
  }
}

// ---------------- additive attention: scores + softmax + context ----------------
__device__ __forceinline__ float fast_tanh(float x) {
  x = fminf(fmaxf(x, -15.0f), 15.0f);
  float e = __expf(2.0f * x);
  return (e - 1.0f) / (e + 1.0f);
}

__global__ __launch_bounds__(256) void attn_kernel(
    const float* __restrict__ Qp, const float* __restrict__ Kp,
    const float* __restrict__ V, const unsigned char* __restrict__ mask,
    const float* __restrict__ vg, float* __restrict__ distr,
    float* __restrict__ ctx, int S)
{
  __shared__ float qrow[D_MODELC];
  __shared__ float vv[D_MODELC];
  __shared__ float sc[256];
  __shared__ float red[256];
  int tid = threadIdx.x;
  int row = blockIdx.x;              // b*T + t
  int b = row / TTC;
  for (int d = tid; d < D_MODELC; d += 256) {
    qrow[d] = Qp[(size_t)row * D_MODELC + d];
    vv[d]   = vg[d];
  }
  __syncthreads();
  float score = -1e9f;
  if (tid < S) {
    const float* kp = Kp + ((size_t)b * S + tid) * D_MODELC;
    float acc = 0.0f;
    for (int d = 0; d < D_MODELC; ++d)
      acc += fast_tanh(qrow[d] + kp[d]) * vv[d];
    score = mask[b * S + tid] ? acc : -1e9f;
  }
  red[tid] = score; __syncthreads();
  for (int off = 128; off > 0; off >>= 1) {
    if (tid < off) red[tid] = fmaxf(red[tid], red[tid + off]);
    __syncthreads();
  }
  float mx = red[0]; __syncthreads();
  float e = __expf(score - mx);      // masked / tid>=S entries underflow to 0
  red[tid] = e; __syncthreads();
  for (int off = 128; off > 0; off >>= 1) {
    if (tid < off) red[tid] += red[tid + off];
    __syncthreads();
  }
  float a = e / red[0];
  sc[tid] = a;
  __syncthreads();
  if (tid < S) distr[(size_t)row * S + tid] = a;
  for (int d = tid; d < D_MODELC; d += 256) {
    float c = 0.0f;
    for (int s = 0; s < S; ++s)
      c += sc[s] * V[((size_t)b * S + s) * D_MODELC + d];
    ctx[(size_t)row * D_MODELC + d] = c;
  }
}

// ---------------- mixture head: softmax([Mnlg,cq,ca,cp] @ Wm + bm) ----------------
__global__ __launch_bounds__(256) void mixture_kernel(
    const float* __restrict__ Mnlg, const float* __restrict__ cq,
    const float* __restrict__ ca, const float* __restrict__ cp,
    const float* __restrict__ Wm, const float* __restrict__ bm,
    float* __restrict__ lambdas)
{
  __shared__ float red[256][4];
  int tid = threadIdx.x, row = blockIdx.x;
  float acc[4] = {0.f, 0.f, 0.f, 0.f};
  for (int i = tid; i < 4 * D_MODELC; i += 256) {
    int seg = i >> 9;
    int d = i & 511;
    float x = (seg == 0) ? Mnlg[(size_t)row * D_MODELC + d]
            : (seg == 1) ? cq[(size_t)row * D_MODELC + d]
            : (seg == 2) ? ca[(size_t)row * D_MODELC + d]
                         : cp[(size_t)row * D_MODELC + d];
    #pragma unroll
    for (int j = 0; j < 4; ++j) acc[j] += x * Wm[(size_t)i * 4 + j];
  }
  for (int j = 0; j < 4; ++j) red[tid][j] = acc[j];
  __syncthreads();
  for (int off = 128; off > 0; off >>= 1) {
    if (tid < off)
      for (int j = 0; j < 4; ++j) red[tid][j] += red[tid + off][j];
    __syncthreads();
  }
  if (tid == 0) {
    float l[4], mx = -1e30f, sm = 0.0f;
    for (int j = 0; j < 4; ++j) { l[j] = red[0][j] + bm[j]; mx = fmaxf(mx, l[j]); }
    for (int j = 0; j < 4; ++j) { l[j] = __expf(l[j] - mx); sm += l[j]; }
    for (int j = 0; j < 4; ++j) lambdas[(size_t)row * 4 + j] = l[j] / sm;
  }
}

// ---------------- in-place vocab softmax * lam_v, zero cols [30000,32000) ----------------
__global__ __launch_bounds__(256) void vocab_softmax_kernel(
    float* __restrict__ out, const unsigned char* __restrict__ special,
    const float* __restrict__ lambdas)
{
  __shared__ float red[256];
  int tid = threadIdx.x, row = blockIdx.x;
  float* p = out + (size_t)row * D_EXTC;
  float mx = -1e30f;
  for (int i = tid; i < D_FIXEDC; i += 256) {
    float x = special[i] ? p[i] : -1e9f;
    mx = fmaxf(mx, x);
  }
  red[tid] = mx; __syncthreads();
  for (int off = 128; off > 0; off >>= 1) {
    if (tid < off) red[tid] = fmaxf(red[tid], red[tid + off]);
    __syncthreads();
  }
  mx = red[0]; __syncthreads();
  float sm = 0.0f;
  for (int i = tid; i < D_FIXEDC; i += 256) {
    float x = special[i] ? p[i] : -1e9f;
    sm += __expf(x - mx);
  }
  red[tid] = sm; __syncthreads();
  for (int off = 128; off > 0; off >>= 1) {
    if (tid < off) red[tid] += red[tid + off];
    __syncthreads();
  }
  float scale = lambdas[(size_t)row * 4 + 0] / red[0];
  for (int i = tid; i < D_FIXEDC; i += 256) {
    float x = special[i] ? p[i] : -1e9f;
    p[i] = __expf(x - mx) * scale;
  }
  for (int i = D_FIXEDC + tid; i < D_EXTC; i += 256) p[i] = 0.0f;
}

// ---------------- scatter-add copy distributions into extended vocab ----------------
__global__ void scatter_kernel(
    float* __restrict__ out, const int* __restrict__ src_ext,
    const float* __restrict__ pd, const float* __restrict__ qd,
    const float* __restrict__ qad, const float* __restrict__ lambdas)
{
  const int STOT = LPC + JQC + NAC;   // 336
  int row = blockIdx.x;
  int b = row / TTC;
  int i = threadIdx.x;
  if (i >= STOT) return;
  float val;
  if (i < LPC)            val = pd[(size_t)row * LPC + i] * lambdas[(size_t)row * 4 + 3];
  else if (i < LPC + JQC) val = qd[(size_t)row * JQC + (i - LPC)] * lambdas[(size_t)row * 4 + 1];
  else                    val = qad[(size_t)row * NAC + (i - LPC - JQC)] * lambdas[(size_t)row * 4 + 2];
  int idx = src_ext[(size_t)b * STOT + i];
  atomicAdd(out + (size_t)row * D_EXTC + idx, val);
}

// ---------------- host orchestration ----------------
extern "C" void kernel_launch(void* const* d_in, const int* in_sizes, int n_in,
                              void* d_out, int out_size, void* d_ws, size_t ws_size,
                              hipStream_t stream) {
  (void)in_sizes; (void)n_in; (void)out_size; (void)ws_size;
  const float* Mp   = (const float*)d_in[0];
  const float* Mq   = (const float*)d_in[1];
  const float* Mqa  = (const float*)d_in[2];
  const float* Mnlg = (const float*)d_in[3];
  const unsigned char* mask_p  = (const unsigned char*)d_in[4];
  const unsigned char* mask_q  = (const unsigned char*)d_in[5];
  const unsigned char* mask_qa = (const unsigned char*)d_in[6];
  const int* src_ext = (const int*)d_in[7];
  const unsigned char* special = (const unsigned char*)d_in[8];
  // d_in[9] = d_ext_vocab scalar (hardcoded)
  const float* Wk_q = (const float*)d_in[10]; const float* bk_q = (const float*)d_in[11];
  const float* Wq_q = (const float*)d_in[12]; const float* bq_q = (const float*)d_in[13];
  const float* v_q  = (const float*)d_in[14];
  const float* Wk_a = (const float*)d_in[15]; const float* bk_a = (const float*)d_in[16];
  const float* Wq_a = (const float*)d_in[17]; const float* bq_a = (const float*)d_in[18];
  const float* v_a  = (const float*)d_in[19];
  const float* Wk_p = (const float*)d_in[20]; const float* bk_p = (const float*)d_in[21];
  const float* Wq_p = (const float*)d_in[22]; const float* bq_p = (const float*)d_in[23];
  const float* v_p  = (const float*)d_in[24];
  const float* W1 = (const float*)d_in[25]; const float* b1 = (const float*)d_in[26];
  const float* W2 = (const float*)d_in[27];
  const float* Wm = (const float*)d_in[28]; const float* bm = (const float*)d_in[29];

  float* out = (float*)d_out;
  float* lambdas = out + (size_t)ROWSC * D_EXTC;

  char* w = (char*)d_ws;
  auto alloc = [&](size_t bytes) -> void* {
    void* p = (void*)w;
    w += (bytes + 255) & ~(size_t)255;
    return p;
  };
  // transposed f16 weights (K-contiguous rows)
  half_t* tWkq = (half_t*)alloc((size_t)512 * 512 * 2);
  half_t* tWqq = (half_t*)alloc((size_t)512 * 512 * 2);
  half_t* tWka = (half_t*)alloc((size_t)512 * 512 * 2);
  half_t* tWqa = (half_t*)alloc((size_t)512 * 512 * 2);
  half_t* tWkp = (half_t*)alloc((size_t)512 * 512 * 2);
  half_t* tWqp = (half_t*)alloc((size_t)512 * 512 * 2);
  half_t* tW1  = (half_t*)alloc((size_t)384 * 512 * 2);            // [Npad=384][K=512]
  half_t* tW2  = (half_t*)alloc((size_t)D_FIXEDP * D_EMBP * 2);    // [30080][320]
  // f16 activations
  half_t* hMp   = (half_t*)alloc((size_t)2048 * 512 * 2);
  half_t* hMq   = (half_t*)alloc((size_t)384 * 512 * 2);
  half_t* hMqa  = (half_t*)alloc((size_t)256 * 512 * 2);
  half_t* hMnlg = (half_t*)alloc((size_t)384 * 512 * 2);
  half_t* hHid  = (half_t*)alloc((size_t)384 * D_EMBP * 2);        // [384][320], zero-padded
  // f32 intermediates
  float* Kp_p = (float*)alloc((size_t)2048 * 512 * 4);
  float* Kp_q = (float*)alloc((size_t)384 * 512 * 4);
  float* Kp_a = (float*)alloc((size_t)256 * 512 * 4);
  float* Qp_q = (float*)alloc((size_t)384 * 512 * 4);
  float* Qp_a = (float*)alloc((size_t)384 * 512 * 4);
  float* Qp_p = (float*)alloc((size_t)384 * 512 * 4);
  float* hidf = (float*)alloc((size_t)384 * 300 * 4);
  float* ctx_q = (float*)alloc((size_t)384 * 512 * 4);
  float* ctx_a = (float*)alloc((size_t)384 * 512 * 4);
  float* ctx_p = (float*)alloc((size_t)384 * 512 * 4);
  float* pdist  = (float*)alloc((size_t)384 * 256 * 4);
  float* qdist  = (float*)alloc((size_t)384 * 48 * 4);
  float* qadist = (float*)alloc((size_t)384 * 32 * 4);

  auto cvt = [&](const float* s, half_t* d, int n) {
    cvt_f32_f16<<<(n + 255) / 256, 256, 0, stream>>>(s, d, n);
  };
  auto tpose = [&](const float* s, half_t* d, int K, int N, int Kpad, int Npad) {
    dim3 g(Npad / 32, Kpad / 32);
    transpose_cvt<<<g, dim3(32, 8), 0, stream>>>(s, d, K, N, Kpad);
  };
  auto gemm = [&](const half_t* A, const half_t* Bt, const float* bias, float* C,
                  int M, int lda, int ldb, int Kpad, int N, int ldc) {
    dim3 g(M / GBM, (N + GBN - 1) / GBN);
    gemm_f16_wmma<<<g, 256, 0, stream>>>(A, Bt, bias, C, lda, ldb, Kpad, N, ldc);
  };

  // stage weights (transpose+pad+f16) and activations (f16)
  tpose(Wk_q, tWkq, 512, 512, 512, 512);
  tpose(Wq_q, tWqq, 512, 512, 512, 512);
  tpose(Wk_a, tWka, 512, 512, 512, 512);
  tpose(Wq_a, tWqa, 512, 512, 512, 512);
  tpose(Wk_p, tWkp, 512, 512, 512, 512);
  tpose(Wq_p, tWqp, 512, 512, 512, 512);
  tpose(W1, tW1, 512, D_EMBC, 512, 384);
  tpose(W2, tW2, D_EMBC, D_FIXEDC, D_EMBP, D_FIXEDP);
  cvt(Mp, hMp, 2048 * 512);    cvt(Mq, hMq, 384 * 512);
  cvt(Mqa, hMqa, 256 * 512);   cvt(Mnlg, hMnlg, 384 * 512);

  // projections (WMMA GEMMs)
  gemm(hMq,   tWkq, bk_q, Kp_q, 384,  512, 512, 512, 512, 512);
  gemm(hMqa,  tWka, bk_a, Kp_a, 256,  512, 512, 512, 512, 512);
  gemm(hMp,   tWkp, bk_p, Kp_p, 2048, 512, 512, 512, 512, 512);
  gemm(hMnlg, tWqq, bq_q, Qp_q, 384,  512, 512, 512, 512, 512);
  gemm(hMnlg, tWqa, bq_a, Qp_a, 384,  512, 512, 512, 512, 512);
  gemm(hMnlg, tWqp, bq_p, Qp_p, 384,  512, 512, 512, 512, 512);

  // vocab generator: hidden then logits fused straight into d_out (ldc = 32000)
  gemm(hMnlg, tW1, b1, hidf, 384, 512, 512, 512, D_EMBC, D_EMBC);
  cvt_pad_rows<<<(384 * D_EMBP + 255) / 256, 256, 0, stream>>>(
      hidf, hHid, D_EMBC, D_EMBP, 384 * D_EMBP);
  gemm(hHid, tW2, nullptr, out, 384, D_EMBP, D_EMBP, D_EMBP, D_FIXEDC, D_EXTC);

  // additive attentions
  attn_kernel<<<ROWSC, 256, 0, stream>>>(Qp_q, Kp_q, Mq,  mask_q,  v_q, qdist,  ctx_q, JQC);
  attn_kernel<<<ROWSC, 256, 0, stream>>>(Qp_a, Kp_a, Mqa, mask_qa, v_a, qadist, ctx_a, NAC);
  attn_kernel<<<ROWSC, 256, 0, stream>>>(Qp_p, Kp_p, Mp,  mask_p,  v_p, pdist,  ctx_p, LPC);

  // mixture weights -> d_out tail
  mixture_kernel<<<ROWSC, 256, 0, stream>>>(Mnlg, ctx_q, ctx_a, ctx_p, Wm, bm, lambdas);

  // fixed-vocab softmax * lam_v in place, zero extension columns
  vocab_softmax_kernel<<<ROWSC, 256, 0, stream>>>(out, special, lambdas);

  // scatter-add copy distributions on top
  scatter_kernel<<<ROWSC, 352, 0, stream>>>(out, src_ext, pdist, qdist, qadist, lambdas);
}
